// rnn_65807488909673
// MI455X (gfx1250) — matrix operand order repaired and checked by reading, not compiled, not run on hardware
//
#include <hip/hip_runtime.h>
#include <hip/hip_bf16.h>
#include <cstdint>

typedef __attribute__((ext_vector_type(16))) __bf16 v16bf;
typedef __attribute__((ext_vector_type(8)))  __bf16 v8bf;
typedef __attribute__((ext_vector_type(8)))  float  v8f;

static constexpr float TAUF = 0.25f;
static constexpr int Bn = 2048, On = 512, Hn = 1024, Pn = 1024, Cn = 512, Tn = 12;
static constexpr int KP = Hn + Pn + Cn; // 2560 = concatenated [ah|ap|ac] width

__device__ __forceinline__ float sigmoidf_(float x) {
  return 1.0f / (1.0f + __expf(-x));
}

// Build one WMMA operand fragment (v16bf) from two 16B-aligned 8-element loads.
// A operand: off2 = 16 (K swizzle blocks of 8 per lane-half, +16 apart)
// B operand: off2 = 16 (same blocks-of-8 K order as A)
__device__ __forceinline__ v16bf load_frag(const __bf16* __restrict__ p,
                                           int off2) {
  v8bf lo = *(const v8bf*)(p);
  v8bf hi = *(const v8bf*)(p + off2);
  return __builtin_shufflevector(lo, hi, 0, 1, 2, 3, 4, 5, 6, 7, 8, 9, 10, 11,
                                 12, 13, 14, 15);
}

__device__ __forceinline__ void mma8(v8f (&acc)[4][2], const v16bf (&a)[4],
                                     const v16bf (&b)[2]) {
#pragma unroll
  for (int mi = 0; mi < 4; ++mi)
#pragma unroll
    for (int ni = 0; ni < 2; ++ni)
      acc[mi][ni] = __builtin_amdgcn_wmma_f32_16x16x32_bf16(
          false, a[mi], false, b[ni], (short)0, acc[mi][ni], false, false);
}

// ---------------------------------------------------------------------------
// Fused bf16 WMMA GEMM tile:  Z[128,128] = A[M, kA0:kA0+K] * Bt[N,K]^T
// 8 waves (2 Mwaves x 4 Nwaves); wave tile 64x32 = 4x2 v_wmma_f32_16x16x32_bf16.
// Double-buffered fragment pipeline, K unrolled by 64 (K % 64 == 0 required;
// tail over-fetch of 32 columns stays inside the padded workspace).
// MODE 0: outF32[row*N+col] = z + bias[col]                   (oh GEMM)
// MODE 1: p-state leaky update + sigmoid -> outF32 + outBF16  (p step)
// MODE 2: c-state leaky update + sigmoid -> outBF16           (c step)
// ---------------------------------------------------------------------------
template <int MODE>
__device__ __forceinline__ void gemm_tile(
    const __bf16* __restrict__ A, int lda, int kA0,
    const __bf16* __restrict__ Bt, int N, int K,
    const float* __restrict__ bias, float* __restrict__ state,
    float* __restrict__ outF32, __bf16* __restrict__ outBF16, int ldx, int bm,
    int bn) {
  const int tid  = threadIdx.x;
  const int wid  = tid >> 5;
  const int lane = tid & 31;
  const int half = lane >> 4;
  const int l15  = lane & 15;

  const int block_m = bm * 128;
  const int block_n = bn * 128;
  const int wave_m  = (wid >> 2) * 64;
  const int wave_n  = (wid & 3) * 32;

  v8f acc[4][2] = {};

  // A layout (16-bit A 16x32, ISA 7.12.2): lanes 0-15 hold K 0-7/16-23,
  // lanes 16-31 hold K 8-15/24-31, row = l15.
  const __bf16* Abase =
      A + (size_t)(block_m + wave_m + l15) * lda + kA0 + half * 8;
  // B layout (16-bit B 32x16): lane = output column; lanes 0-15 hold K 0-7/16-23,
  // lanes 16-31 hold K 8-15/24-31 (same K order as A).
  const __bf16* Bbase = Bt + (size_t)(block_n + wave_n + l15) * K + half * 8;

  v16bf aC[4], bC[2], aN[4], bN[2];
#pragma unroll
  for (int mi = 0; mi < 4; ++mi)
    aC[mi] = load_frag(Abase + (size_t)(mi * 16) * lda, 16);
#pragma unroll
  for (int ni = 0; ni < 2; ++ni)
    bC[ni] = load_frag(Bbase + (size_t)(ni * 16) * K, 16);

  for (int k0 = 0; k0 < K; k0 += 64) {
#pragma unroll
    for (int mi = 0; mi < 4; ++mi)
      aN[mi] = load_frag(Abase + (size_t)(mi * 16) * lda + k0 + 32, 16);
#pragma unroll
    for (int ni = 0; ni < 2; ++ni)
      bN[ni] = load_frag(Bbase + (size_t)(ni * 16) * K + k0 + 32, 16);
    // locality 3 -> WGP-scope prefetch (pull into near cache; SYS/DEV-scope
    // prefetch would stop at GL2 where this working set already resides).
    __builtin_prefetch(Abase + k0 + 512, 0, 3);
    __builtin_prefetch(Bbase + k0 + 512, 0, 3);
    mma8(acc, aC, bC);
#pragma unroll
    for (int mi = 0; mi < 4; ++mi)
      aC[mi] = load_frag(Abase + (size_t)(mi * 16) * lda + k0 + 64, 16);
#pragma unroll
    for (int ni = 0; ni < 2; ++ni)
      bC[ni] = load_frag(Bbase + (size_t)(ni * 16) * K + k0 + 64, 16);
    mma8(acc, aN, bN);
  }

  // Epilogue. C/D layout: VGPR i -> M = i (lanes 0-15) / M = 8+i (lanes 16-31).
#pragma unroll
  for (int mi = 0; mi < 4; ++mi) {
#pragma unroll
    for (int ni = 0; ni < 2; ++ni) {
#pragma unroll
      for (int i = 0; i < 8; ++i) {
        const int row = block_m + wave_m + mi * 16 + half * 8 + i;
        const int col = block_n + wave_n + ni * 16 + l15;
        const float z = acc[mi][ni][i];
        if constexpr (MODE == 0) {
          outF32[(size_t)row * N + col] = z + bias[col];
        } else {
          const size_t idx = (size_t)row * N + col;
          const float s_new =
              TAUF * (z + bias[col]) + (1.0f - TAUF) * state[idx];
          state[idx] = s_new;
          const float a_new = sigmoidf_(s_new);
          if constexpr (MODE == 1) outF32[idx] = a_new;
          outBF16[(size_t)row * ldx + col] = (__bf16)a_new;
        }
      }
    }
  }
}

// One fused recurrent step: grid = (16, 16)
//   blockIdx.y in [0,8)  : p-GEMM  (M=2048, N=1024, K=2560)
//   blockIdx.y in [8,12) : c-GEMM  (M=2048, N=512,  K=1024)
//   blockIdx.y in [12,16): ah_{t+1} = sigmoid(oh * scale) -> Xnext[:, 0:H)
__global__ __launch_bounds__(256) void step_kernel(
    const __bf16* __restrict__ Xcur, const __bf16* __restrict__ WTp,
    const __bf16* __restrict__ WTc, const float* __restrict__ bias_p,
    const float* __restrict__ bias_c, float* __restrict__ pstate,
    float* __restrict__ cstate, const float* __restrict__ oh, float ah_scale,
    float* __restrict__ outP, __bf16* __restrict__ Xnxt) {
  const int by = blockIdx.y;
  if (by < 8) {
    gemm_tile<1>(Xcur, KP, 0, WTp, Pn, KP, bias_p, pstate, outP, Xnxt + Hn, KP,
                 blockIdx.x, by);
  } else if (by < 12) {
    gemm_tile<2>(Xcur, KP, Hn, WTc, Cn, Pn, bias_c, cstate, nullptr,
                 Xnxt + Hn + Pn, KP, blockIdx.x, by - 8);
  } else {
    const int blk = (by - 12) * 16 + blockIdx.x;  // 0..63
    const int per_blk = (Bn * Hn) / 64;           // 32768, multiple of 1024
    const int end = blk * per_blk + per_blk;
    for (int i = blk * per_blk + (int)threadIdx.x * 4; i < end; i += 256 * 4) {
      float4 v = *(const float4*)(oh + i);
      const int row = i >> 10;          // Hn = 1024
      const int col = i & (Hn - 1);
      __bf16* d = Xnxt + (size_t)row * KP + col;
      d[0] = (__bf16)sigmoidf_(v.x * ah_scale);
      d[1] = (__bf16)sigmoidf_(v.y * ah_scale);
      d[2] = (__bf16)sigmoidf_(v.z * ah_scale);
      d[3] = (__bf16)sigmoidf_(v.w * ah_scale);
    }
  }
}

// oh = inputs @ w_oh + bias_h   (grid (16, 8))
__global__ __launch_bounds__(256) void gemm_oh_kernel(
    const __bf16* __restrict__ Xin, const __bf16* __restrict__ WToh,
    const float* __restrict__ bias_h, float* __restrict__ oh) {
  gemm_tile<0>(Xin, On, 0, WToh, Hn, On, bias_h, nullptr, oh, nullptr, 0,
               blockIdx.x, blockIdx.y);
}

// --------------------------- setup kernels ---------------------------------
__global__ void fill_bf16_kernel(__bf16* p, float v, int n) {
  int i = blockIdx.x * blockDim.x + threadIdx.x;
  if (i < n) p[i] = (__bf16)v;
}
__global__ void fill_f32_kernel(float* p, float v, int n) {
  int i = blockIdx.x * blockDim.x + threadIdx.x;
  if (i < n) p[i] = v;
}
__global__ void cast_f32_bf16_kernel(const float* __restrict__ s,
                                     __bf16* __restrict__ d, int n) {
  int i = blockIdx.x * blockDim.x + threadIdx.x;
  if (i < n) d[i] = (__bf16)s[i];
}
// src[K][N] fp32 -> dst[N][K] bf16  (transpose + cast)
__global__ void transpose_cast_kernel(const float* __restrict__ src,
                                      __bf16* __restrict__ dst, int K, int N) {
  int i = blockIdx.x * blockDim.x + threadIdx.x;
  if (i >= N * K) return;
  int n = i / K, k = i % K;
  dst[i] = (__bf16)src[(size_t)k * N + n];
}
// Stacked, transposed, diag-zeroed p weights: dst[P][KP]
__global__ void build_wtp_kernel(const float* __restrict__ w_hp,
                                 const float* __restrict__ w_pp,
                                 const float* __restrict__ w_cp,
                                 __bf16* __restrict__ dst) {
  int i = blockIdx.x * blockDim.x + threadIdx.x;
  if (i >= Pn * KP) return;
  int n = i / KP, k = i % KP;
  float v;
  if (k < Hn) {
    v = w_hp[(size_t)k * Pn + n];
  } else if (k < Hn + Pn) {
    int kk = k - Hn;
    v = (kk == n) ? 0.0f : w_pp[(size_t)kk * Pn + n];
  } else {
    int kk = k - Hn - Pn;
    v = w_cp[(size_t)kk * Pn + n];
  }
  dst[i] = (__bf16)v;
}

// ------------------------------- host side ---------------------------------
extern "C" void kernel_launch(void* const* d_in, const int* in_sizes, int n_in,
                              void* d_out, int out_size, void* d_ws,
                              size_t ws_size, hipStream_t stream) {
  (void)in_sizes; (void)n_in; (void)out_size; (void)ws_size;
  const float* inputs = (const float*)d_in[0];  // [B,O]
  const float* w_oh   = (const float*)d_in[1];  // [O,H]
  const float* w_hp   = (const float*)d_in[2];  // [H,P]
  const float* w_pp   = (const float*)d_in[3];  // [P,P]
  const float* w_pc   = (const float*)d_in[4];  // [P,C]
  const float* w_cp   = (const float*)d_in[5];  // [C,P]
  const float* bias_h = (const float*)d_in[6];
  const float* bias_p = (const float*)d_in[7];
  const float* bias_c = (const float*)d_in[8];
  float* out = (float*)d_out;  // [T,B,P]

  // Workspace layout (256B aligned; +4KB guard for pipeline tail over-fetch)
  char* ws = (char*)d_ws;
  auto alloc = [&](size_t bytes) {
    char* p = ws;
    ws += (bytes + 255) & ~size_t(255);
    return p;
  };
  float*  oh     = (float*)alloc((size_t)Bn * Hn * 4);   // 8 MB
  float*  pstate = (float*)alloc((size_t)Bn * Pn * 4);   // 8 MB
  float*  cstate = (float*)alloc((size_t)Bn * Cn * 4);   // 4 MB
  __bf16* X0     = (__bf16*)alloc((size_t)Bn * KP * 2);  // 10 MB
  __bf16* X1     = (__bf16*)alloc((size_t)Bn * KP * 2);  // 10 MB
  __bf16* Xin    = (__bf16*)alloc((size_t)Bn * On * 2);  // 2 MB
  __bf16* WToh   = (__bf16*)alloc((size_t)Hn * On * 2);  // 1 MB
  __bf16* WTp    = (__bf16*)alloc((size_t)Pn * KP * 2);  // 5 MB
  __bf16* WTc    = (__bf16*)alloc((size_t)Cn * Pn * 2);  // 1 MB
  (void)alloc(4096);                                     // guard pad

  const int TB = 256;
  auto blocks = [&](int n) { return (n + TB - 1) / TB; };

  // ---- one-time setup (deterministic, re-done each call) ----
  fill_bf16_kernel<<<blocks(Bn * KP), TB, 0, stream>>>(X0, 0.5f, Bn * KP);
  fill_f32_kernel<<<blocks(Bn * Pn), TB, 0, stream>>>(pstate, 0.0f, Bn * Pn);
  fill_f32_kernel<<<blocks(Bn * Cn), TB, 0, stream>>>(cstate, 0.0f, Bn * Cn);
  cast_f32_bf16_kernel<<<blocks(Bn * On), TB, 0, stream>>>(inputs, Xin,
                                                           Bn * On);
  transpose_cast_kernel<<<blocks(Hn * On), TB, 0, stream>>>(w_oh, WToh, On, Hn);
  transpose_cast_kernel<<<blocks(Cn * Pn), TB, 0, stream>>>(w_pc, WTc, Pn, Cn);
  build_wtp_kernel<<<blocks(Pn * KP), TB, 0, stream>>>(w_hp, w_pp, w_cp, WTp);

  // oh = inputs @ w_oh + bias_h
  {
    dim3 grid(Bn / 128, Hn / 128);
    gemm_oh_kernel<<<grid, 256, 0, stream>>>(Xin, WToh, bias_h, oh);
  }

  // ---- 12 recurrent steps, one fused launch each ----
  __bf16* Xcur = X0;
  __bf16* Xnxt = X1;
  float pw = 1.0f;
  for (int t = 0; t < Tn; ++t) {
    pw *= (1.0f - TAUF);
    dim3 grid(16, 16);
    step_kernel<<<grid, 256, 0, stream>>>(
        Xcur, WTp, WTc, bias_p, bias_c, pstate, cstate, oh, 1.0f - pw,
        out + (size_t)t * Bn * Pn, Xnxt);
    __bf16* tmp = Xcur; Xcur = Xnxt; Xnxt = tmp;
  }
}
